// TopKSAE_53618371723771
// MI455X (gfx1250) — matrix operand order repaired and checked by reading, not compiled, not run on hardware
//
#include <hip/hip_runtime.h>
#include <hip/hip_bf16.h>

// ---------------------------------------------------------------------------
// Types for CDNA5 WMMA
// ---------------------------------------------------------------------------
typedef __attribute__((ext_vector_type(16))) __bf16 v16bf;
typedef __attribute__((ext_vector_type(8)))  float  v8f;

union BFrag16 {            // 16 bf16 = 32 bytes = two uint4 LDS/b128 loads
    uint4  q[2];
    v16bf  v;
};

#define SAE_N      8192
#define SAE_DM     768
#define SAE_F      24576
#define SAE_K      32

// ---------------------------------------------------------------------------
// 0) f32 -> bf16 conversion (grid-stride)
// ---------------------------------------------------------------------------
__global__ void sae_cvt_bf16(const float* __restrict__ src,
                             __hip_bfloat16* __restrict__ dst, int n) {
    for (int i = blockIdx.x * 256 + threadIdx.x; i < n; i += gridDim.x * 256)
        dst[i] = __float2bfloat16(src[i]);
}

// ---------------------------------------------------------------------------
// 1) W_dec [768, 24576] -> W_decT [24576, 768]  (coalesced decode gathers)
// ---------------------------------------------------------------------------
__global__ __launch_bounds__(256) void sae_transpose_wdec(
        const float* __restrict__ Wd, float* __restrict__ WdT) {
    __shared__ float tile[32][33];
    const int tx = threadIdx.x & 31;
    const int ty = threadIdx.x >> 5;        // 0..7
    const int f0 = blockIdx.x * 32;         // over 24576
    const int d0 = blockIdx.y * 32;         // over 768
#pragma unroll
    for (int j = 0; j < 32; j += 8)
        tile[ty + j][tx] = Wd[(size_t)(d0 + ty + j) * SAE_F + f0 + tx];
    __syncthreads();
#pragma unroll
    for (int j = 0; j < 32; j += 8)
        WdT[(size_t)(f0 + ty + j) * SAE_DM + d0 + tx] = tile[tx][ty + j];
}

// ---------------------------------------------------------------------------
// 2) Encoder GEMM: z[n,f] = sum_d xb[n,d]*wb[f,d] + b_enc[f]
//    bf16 inputs, f32 accumulate via v_wmma_f32_16x16x32_bf16.
//    Block tile 128x128, K-chunk 32. 8 waves; wave tile 32(M) x 64(N).
// ---------------------------------------------------------------------------
__global__ __launch_bounds__(256) void sae_encode_gemm(
        const __hip_bfloat16* __restrict__ xb,   // [8192, 768]
        const __hip_bfloat16* __restrict__ wb,   // [24576, 768]
        const float* __restrict__ b_enc,         // [24576]
        float* __restrict__ z_out)               // [8192, 24576]
{
    // 128 rows x 32 bf16 = 8 KB each; stored as uint4 (8 bf16 per uint4)
    __shared__ uint4 Ash[128 * 4];
    __shared__ uint4 Bsh[128 * 4];

    const int tid  = threadIdx.x;
    const int lane = tid & 31;
    const int wave = tid >> 5;
    const int wm   = wave & 3;        // 4 waves along M: 32 rows each
    const int wn   = wave >> 2;       // 2 waves along N: 64 cols each
    const int l16  = lane & 15;
    const int half = lane >> 4;

    const int m0 = blockIdx.y * 128;
    const int n0 = blockIdx.x * 128;

    const uint4* xq = (const uint4*)xb;   // row stride = 768/8 = 96 uint4
    const uint4* wq = (const uint4*)wb;

    v8f acc[2][4];
#pragma unroll
    for (int i = 0; i < 2; ++i)
#pragma unroll
        for (int j = 0; j < 4; ++j)
            acc[i][j] = (v8f){0.f, 0.f, 0.f, 0.f, 0.f, 0.f, 0.f, 0.f};

    for (int kc = 0; kc < SAE_DM; kc += 32) {
        const int kq = kc >> 3;           // uint4 offset inside a row
#pragma unroll
        for (int i = 0; i < 2; ++i) {     // 512 uint4 per tile / 256 threads
            const int idx = tid + i * 256;
            const int row = idx >> 2;
            const int q   = idx & 3;
            Ash[idx] = xq[(size_t)(m0 + row) * 96 + kq + q];
            Bsh[idx] = wq[(size_t)(n0 + row) * 96 + kq + q];
        }
        __syncthreads();

        // Build fragments (ISA 7.12.2 layouts):
        // A 16x32: lane<16 -> K{0..7,16..23}; lane>=16 -> K{8..15,24..31}
        // B 32x16: lane<16 -> K{0..7,16..23}; lane>=16 -> K{8..15,24..31}
        BFrag16 a[2], b[4];
#pragma unroll
        for (int mt = 0; mt < 2; ++mt) {
            const int r = wm * 32 + mt * 16 + l16;
            a[mt].q[0] = Ash[r * 4 + half];
            a[mt].q[1] = Ash[r * 4 + 2 + half];
        }
#pragma unroll
        for (int nt = 0; nt < 4; ++nt) {
            const int r = wn * 64 + nt * 16 + l16;
            b[nt].q[0] = Bsh[r * 4 + half];
            b[nt].q[1] = Bsh[r * 4 + 2 + half];
        }

#pragma unroll
        for (int mt = 0; mt < 2; ++mt)
#pragma unroll
            for (int nt = 0; nt < 4; ++nt)
                acc[mt][nt] = __builtin_amdgcn_wmma_f32_16x16x32_bf16(
                    false, a[mt].v, false, b[nt].v,
                    (short)0, acc[mt][nt], false, false);
        __syncthreads();
    }

    // Epilogue: fuse b_enc, write z. C layout: VGPR r -> M=r(+8 for hi lanes),
    // N = lane&15 within each 16x16 tile.
#pragma unroll
    for (int nt = 0; nt < 4; ++nt) {
        const int col  = n0 + wn * 64 + nt * 16 + l16;
        const float bias = b_enc[col];
#pragma unroll
        for (int mt = 0; mt < 2; ++mt) {
            const int mbase = m0 + wm * 32 + mt * 16 + (half << 3);
#pragma unroll
            for (int r = 0; r < 8; ++r)
                z_out[(size_t)(mbase + r) * SAE_F + col] = acc[mt][nt][r] + bias;
        }
    }
}

// ---------------------------------------------------------------------------
// 3) Per-row top-K (K=32) radix select + ReLU filter.
//    One block per row; row staged in 96 KB dynamic LDS (CDNA5 WGP: 320 KB).
// ---------------------------------------------------------------------------
__global__ __launch_bounds__(256) void sae_topk(
        float* __restrict__ z,          // in/out sparse region [8192, 24576]
        float* __restrict__ tk_val,     // [8192, 32]
        int*   __restrict__ tk_idx)     // [8192, 32]
{
    extern __shared__ unsigned int keys[];   // 24576 ordered keys
    __shared__ int s_cnt;
    __shared__ int s_sel;

    const int row = blockIdx.x;
    const int tid = threadIdx.x;
    float* zr = z + (size_t)row * SAE_F;

    // monotone float -> uint mapping (larger float => larger key)
    for (int i = tid; i < SAE_F; i += 256) {
        unsigned int u = __float_as_uint(zr[i]);
        keys[i] = (u & 0x80000000u) ? ~u : (u | 0x80000000u);
    }
    if (tid == 0) s_sel = 0;
    __syncthreads();

    // bitwise radix select of the K-th largest key
    unsigned int prefix = 0u, mask = 0u;
    int need = SAE_K;
    for (int bit = 31; bit >= 0; --bit) {
        const unsigned int b = 1u << bit;
        if (tid == 0) s_cnt = 0;
        __syncthreads();
        const unsigned int want = prefix | b;
        const unsigned int m2   = mask | b;
        int local = 0;
        for (int i = tid; i < SAE_F; i += 256)
            local += ((keys[i] & m2) == want);
        atomicAdd(&s_cnt, local);
        __syncthreads();
        const int c = s_cnt;
        if (c >= need) prefix |= b; else need -= c;
        mask |= b;
        __syncthreads();
    }
    const unsigned int thr = prefix;      // key of the K-th largest

    if (tid < SAE_K) {                    // pre-zero compact list (padding)
        tk_val[row * SAE_K + tid] = 0.f;
        tk_idx[row * SAE_K + tid] = 0;
    }
    __syncthreads();

    // filter: keep top-K AND positive (ReLU); rewrite row, emit compact list
    for (int i = tid; i < SAE_F; i += 256) {
        const unsigned int u = keys[i];
        const bool keep = (u >= thr) && (u > 0x80000000u);   // in top-K, > 0
        const float f = __uint_as_float(u & 0x7fffffffu);    // inverse map (>0 case)
        if (keep) {
            const int slot = atomicAdd(&s_sel, 1);
            if (slot < SAE_K) {
                tk_val[row * SAE_K + slot] = f;
                tk_idx[row * SAE_K + slot] = i;
            }
            zr[i] = f;
        } else {
            zr[i] = 0.f;
        }
    }
}

// ---------------------------------------------------------------------------
// 4) Sparse decode: x_hat[n,:] = sum_k val_k * W_decT[idx_k,:] + b_dec
// ---------------------------------------------------------------------------
__global__ __launch_bounds__(256) void sae_decode(
        const float* __restrict__ tk_val,
        const int*   __restrict__ tk_idx,
        const float* __restrict__ WdT,     // [24576, 768] (L2-resident)
        const float* __restrict__ b_dec,   // [768]
        float* __restrict__ xhat)          // [8192, 768]
{
    __shared__ float sval[SAE_K];
    __shared__ int   sidx[SAE_K];
    const int row = blockIdx.x;
    const int tid = threadIdx.x;
    if (tid < SAE_K) {
        sval[tid] = tk_val[row * SAE_K + tid];
        sidx[tid] = tk_idx[row * SAE_K + tid];
    }
    __syncthreads();

    float a0 = b_dec[tid];
    float a1 = b_dec[tid + 256];
    float a2 = b_dec[tid + 512];
#pragma unroll
    for (int k = 0; k < SAE_K; ++k) {
        const float v = sval[k];
        if (v != 0.f) {
            const float* w = WdT + (size_t)sidx[k] * SAE_DM;
            a0 += v * w[tid];
            a1 += v * w[tid + 256];
            a2 += v * w[tid + 512];
        }
    }
    float* o = xhat + (size_t)row * SAE_DM;
    o[tid]       = a0;
    o[tid + 256] = a1;
    o[tid + 512] = a2;
}

// ---------------------------------------------------------------------------
// Launcher
// ---------------------------------------------------------------------------
extern "C" void kernel_launch(void* const* d_in, const int* in_sizes, int n_in,
                              void* d_out, int out_size, void* d_ws, size_t ws_size,
                              hipStream_t stream) {
    (void)in_sizes; (void)n_in; (void)out_size; (void)ws_size;

    const float* x     = (const float*)d_in[0];   // [8192, 768]
    const float* W_enc = (const float*)d_in[1];   // [24576, 768]
    const float* b_enc = (const float*)d_in[2];   // [24576]
    const float* W_dec = (const float*)d_in[3];   // [768, 24576]
    const float* b_dec = (const float*)d_in[4];   // [768]

    float* out    = (float*)d_out;
    float* xhat   = out;                               // [8192, 768]
    float* sparse = out + (size_t)SAE_N * SAE_DM;      // [8192, 24576]

    // workspace layout
    char* ws = (char*)d_ws;
    __hip_bfloat16* xb  = (__hip_bfloat16*)(ws);                    // 12.6 MB
    __hip_bfloat16* wb  = (__hip_bfloat16*)(ws + (size_t)12582912); // 37.7 MB
    float*          WdT = (float*)(ws + (size_t)50331648);          // 75.5 MB
    float*          tkv = (float*)(ws + (size_t)125829120);         //  1  MB
    int*            tki = (int*)  (ws + (size_t)126877696);         //  1  MB

    sae_cvt_bf16<<<dim3(4096), dim3(256), 0, stream>>>(x, xb, SAE_N * SAE_DM);
    sae_cvt_bf16<<<dim3(8192), dim3(256), 0, stream>>>(W_enc, wb, SAE_F * SAE_DM);
    sae_transpose_wdec<<<dim3(SAE_F / 32, SAE_DM / 32), dim3(256), 0, stream>>>(W_dec, WdT);

    sae_encode_gemm<<<dim3(SAE_F / 128, SAE_N / 128), dim3(256), 0, stream>>>(
        xb, wb, b_enc, sparse);

    sae_topk<<<dim3(SAE_N), dim3(256), SAE_F * sizeof(unsigned int), stream>>>(
        sparse, tkv, tki);

    sae_decode<<<dim3(SAE_N), dim3(256), 0, stream>>>(tkv, tki, WdT, b_dec, xhat);
}
